// MultiHeadAttentionBlock_10093173145897
// MI455X (gfx1250) — compile-verified, matrix operand order checked by reading, not run on hardware
//
#include <hip/hip_runtime.h>

typedef __bf16 bf16;
typedef __attribute__((ext_vector_type(16))) __bf16 v16bf;
typedef __attribute__((ext_vector_type(8)))  __bf16 v8bf;
typedef __attribute__((ext_vector_type(8)))  float  v8f;
typedef int v4i __attribute__((vector_size(16))); // b128 payload type for async copies

#define WMMA_BF16(a, b, c) \
  __builtin_amdgcn_wmma_f32_16x16x32_bf16(false, (a), false, (b), (short)0, (c), false, false)

static constexpr int B_ = 2, S_ = 2048, D_ = 1024, H_ = 16, DK_ = 64;
static constexpr int M_ = B_ * S_; // 4096 rows for all projections

// ---- CDNA5 async global->LDS staging (guarded; falls back to sync copy) ----
__device__ inline void async_copy_b128(const void* g, void* l) {
#if __has_builtin(__builtin_amdgcn_global_load_async_to_lds_b128)
  __builtin_amdgcn_global_load_async_to_lds_b128(
      (__attribute__((address_space(1))) v4i*)g,
      (__attribute__((address_space(3))) v4i*)l, 0, 0);
#else
  *(v8bf*)l = *(const v8bf*)g; // sync fallback; __syncthreads orders it
#endif
}

__device__ inline void async_wait_all() {
#if __has_builtin(__builtin_amdgcn_global_load_async_to_lds_b128)
#if __has_builtin(__builtin_amdgcn_s_wait_asynccnt)
  __builtin_amdgcn_s_wait_asynccnt(0);
#else
  asm volatile("s_wait_asynccnt 0x0" ::: "memory");
#endif
#endif
}

// Load one 16x32 bf16 WMMA operand fragment (A-layout; B-operands use the same
// striping with "row" = the N column index). Per ISA 7.12.2: lane l<16 holds
// row l, K={0..7,16..23}; lane l+16 holds row l, K={8..15,24..31}.
__device__ inline v16bf load_frag(const bf16* __restrict__ base, int ld, int lane) {
  const int r = lane & 15, half = lane >> 4;
  const bf16* p = base + r * ld + half * 8;
  v8bf lo = *(const v8bf*)(p);
  v8bf hi = *(const v8bf*)(p + 16);
  v16bf out;
#pragma unroll
  for (int i = 0; i < 8; ++i) { out[i] = lo[i]; out[8 + i] = hi[i]; }
  return out;
}

__global__ void cvt_f32_to_bf16(const float* __restrict__ src, bf16* __restrict__ dst, int n) {
  int i = blockIdx.x * blockDim.x + threadIdx.x;
  int stride = gridDim.x * blockDim.x;
  for (; i < n; i += stride) dst[i] = (bf16)src[i];
}

// C = A[M,K] * W[N,K]^T + bias. mode 0: fp32 out [M,N]; mode 1: bf16 out split
// to heads [B,H,S,DK]. Wave tile 32x64 (2x4 WMMA accums, 8 WMMA / k-step);
// waves 4(M) x 2(N): workgroup tile 128x128.
__global__ __launch_bounds__(256) void gemm_bf16(
    const bf16* __restrict__ A, const bf16* __restrict__ W,
    const float* __restrict__ bias, float* __restrict__ outF,
    bf16* __restrict__ outH, int M, int N, int K, int mode) {
  const int lane = threadIdx.x & 31;
  const int wave = threadIdx.x >> 5;
  const int mw = wave & 3, nw = wave >> 2;
  const int m0 = blockIdx.y * 128 + mw * 32;
  const int n0 = blockIdx.x * 128 + nw * 64;
  v8f c[2][4] = {};
  for (int k0 = 0; k0 < K; k0 += 32) {
    v16bf a0 = load_frag(A + m0 * K + k0, K, lane);
    v16bf a1 = load_frag(A + (m0 + 16) * K + k0, K, lane);
    v16bf b0 = load_frag(W + (n0 + 0) * K + k0, K, lane);
    v16bf b1 = load_frag(W + (n0 + 16) * K + k0, K, lane);
    v16bf b2 = load_frag(W + (n0 + 32) * K + k0, K, lane);
    v16bf b3 = load_frag(W + (n0 + 48) * K + k0, K, lane);
    c[0][0] = WMMA_BF16(a0, b0, c[0][0]);
    c[0][1] = WMMA_BF16(a0, b1, c[0][1]);
    c[0][2] = WMMA_BF16(a0, b2, c[0][2]);
    c[0][3] = WMMA_BF16(a0, b3, c[0][3]);
    c[1][0] = WMMA_BF16(a1, b0, c[1][0]);
    c[1][1] = WMMA_BF16(a1, b1, c[1][1]);
    c[1][2] = WMMA_BF16(a1, b2, c[1][2]);
    c[1][3] = WMMA_BF16(a1, b3, c[1][3]);
  }
  const int half = lane >> 4, nl = lane & 15;
#pragma unroll
  for (int i = 0; i < 2; ++i) {
#pragma unroll
    for (int j = 0; j < 4; ++j) {
      const int n = n0 + j * 16 + nl;
      const float bv = bias[n];
#pragma unroll
      for (int r = 0; r < 8; ++r) {
        const int m = m0 + i * 16 + half * 8 + r;
        const float val = c[i][j][r] + bv;
        if (mode == 0) {
          outF[m * N + n] = val;
        } else {
          const int b = m / S_, s = m % S_;
          const int h = n / DK_, dk = n % DK_;
          outH[((b * H_ + h) * S_ + s) * DK_ + dk] = (bf16)val;
        }
      }
    }
  }
}

// Flash attention: one WG = 128 queries of one (b,h); each wave owns 16
// queries. Keys streamed in blocks of 32; K/V blocks staged into LDS with
// async global->LDS copies shared by all 8 waves; online softmax in registers.
__global__ __launch_bounds__(256) void flash_attn(
    const bf16* __restrict__ Q, const bf16* __restrict__ Kk,
    const bf16* __restrict__ V, const int* __restrict__ mask,
    bf16* __restrict__ X) {
  __shared__ bf16 Ktile[32][DK_];  // K block, row-major [key][dk]
  __shared__ bf16 Vrow[32][DK_];   // V block, row-major [key][dk]
  __shared__ bf16 Vt[DK_][32];     // V block, transposed [dk][key]
  __shared__ bf16 P[8][16][32];    // per-wave probability tiles
  const int lane = threadIdx.x & 31;
  const int wave = threadIdx.x >> 5;
  const int half = lane >> 4, nl = lane & 15;
  const int bh = blockIdx.y;
  const int b = bh / H_, h = bh % H_;
  const int qw = blockIdx.x * 128 + wave * 16;

  const bf16* qbase = Q + (bh * S_ + qw) * DK_;
  const v16bf aq0 = load_frag(qbase, DK_, lane);
  const v16bf aq1 = load_frag(qbase + 32, DK_, lane);

  // per-thread slice for cooperative staging: 32 keys x 8 segments of 16B
  const int ckey = threadIdx.x >> 3;
  const int cseg = (threadIdx.x & 7) * 8;

  v8f o[4] = {};
  float rm[8], rl[8];
#pragma unroll
  for (int r = 0; r < 8; ++r) { rm[r] = -1e30f; rl[r] = 0.0f; }

  for (int key0 = 0; key0 < S_; key0 += 32) {
    __syncthreads(); // previous block's LDS reads complete

    // async stage K and V blocks (one b128 per thread per tile)
    const size_t grow = (size_t)(bh * S_ + key0 + ckey) * DK_ + cseg;
    async_copy_b128(Kk + grow, &Ktile[ckey][cseg]);
    async_copy_b128(V + grow, &Vrow[ckey][cseg]);
    async_wait_all();
    __syncthreads(); // Ktile/Vrow visible to all waves

    // transpose V block in LDS: Vrow[key][dk] -> Vt[dk][key]
    {
      v8bf vv = *(const v8bf*)(&Vrow[ckey][cseg]);
#pragma unroll
      for (int i = 0; i < 8; ++i) Vt[cseg + i][ckey] = vv[i];
    }

    // scores: two 16-key tiles, each = Q(16x64) . K^T(64x16), from LDS
    v8f s[2];
#pragma unroll
    for (int jt = 0; jt < 2; ++jt) {
      v16bf kf0 = load_frag(&Ktile[jt * 16][0], DK_, lane);
      v16bf kf1 = load_frag(&Ktile[jt * 16][32], DK_, lane);
      v8f z = {};
      v8f t = WMMA_BF16(aq0, kf0, z);
      s[jt] = WMMA_BF16(aq1, kf1, t);
    }

    // scale (1/sqrt(64)) + mask
#pragma unroll
    for (int jt = 0; jt < 2; ++jt) {
#pragma unroll
      for (int r = 0; r < 8; ++r) {
        const int qrow = qw + half * 8 + r;
        const int key = key0 + jt * 16 + nl;
        float sv = s[jt][r] * 0.125f;
        if (mask[qrow * S_ + key] == 0) sv = -1e9f;
        s[jt][r] = sv;
      }
    }

    // online softmax (row stats reduced across the 16-lane n-group)
    float alpha[8];
#pragma unroll
    for (int r = 0; r < 8; ++r) {
      float bm = fmaxf(s[0][r], s[1][r]);
#pragma unroll
      for (int off = 1; off < 16; off <<= 1) bm = fmaxf(bm, __shfl_xor(bm, off, 32));
      const float mnew = fmaxf(rm[r], bm);
      alpha[r] = __expf(rm[r] - mnew);
      const float p0 = __expf(s[0][r] - mnew);
      const float p1 = __expf(s[1][r] - mnew);
      rm[r] = mnew;
      float part = p0 + p1;
#pragma unroll
      for (int off = 1; off < 16; off <<= 1) part += __shfl_xor(part, off, 32);
      rl[r] = rl[r] * alpha[r] + part;
      const int mrow = half * 8 + r;
      P[wave][mrow][nl] = (bf16)p0;
      P[wave][mrow][16 + nl] = (bf16)p1;
    }

    // rescale running output
#pragma unroll
    for (int c4 = 0; c4 < 4; ++c4)
#pragma unroll
      for (int r = 0; r < 8; ++r) o[c4][r] *= alpha[r];

    __syncthreads(); // Vt + P visible

    // O += P(16x32) . V(32x64)
    const v16bf ap = load_frag(&P[wave][0][0], 32, lane);
#pragma unroll
    for (int c4 = 0; c4 < 4; ++c4) {
      v16bf bv = load_frag(&Vt[c4 * 16][0], 32, lane);
      o[c4] = WMMA_BF16(ap, bv, o[c4]);
    }
  }

  // normalize, merge heads: X[b, s, h*DK + dk] (bf16 for Wo GEMM input)
#pragma unroll
  for (int r = 0; r < 8; ++r) {
    const float inv = 1.0f / rl[r];
    const int qrow = qw + half * 8 + r;
#pragma unroll
    for (int c4 = 0; c4 < 4; ++c4) {
      const int d = h * DK_ + c4 * 16 + nl;
      X[(b * S_ + qrow) * D_ + d] = (bf16)(o[c4][r] * inv);
    }
  }
}

extern "C" void kernel_launch(void* const* d_in, const int* in_sizes, int n_in,
                              void* d_out, int out_size, void* d_ws, size_t ws_size,
                              hipStream_t stream) {
  (void)in_sizes; (void)n_in; (void)out_size; (void)ws_size;
  const float* q  = (const float*)d_in[0];
  const float* k  = (const float*)d_in[1];
  const float* v  = (const float*)d_in[2];
  const int* mask = (const int*)d_in[3];
  const float* Wq = (const float*)d_in[4];
  const float* bq = (const float*)d_in[5];
  const float* Wk = (const float*)d_in[6];
  const float* bk = (const float*)d_in[7];
  const float* Wv = (const float*)d_in[8];
  const float* bv = (const float*)d_in[9];
  const float* Wo = (const float*)d_in[10];
  const float* bo = (const float*)d_in[11];
  float* out = (float*)d_out;

  char* ws = (char*)d_ws;
  size_t off = 0;
  auto alloc = [&](size_t bytes) -> bf16* {
    bf16* p = (bf16*)(ws + off);
    off += (bytes + 255) & ~size_t(255);
    return p;
  };
  const size_t szXD = (size_t)M_ * D_ * sizeof(bf16); // 8 MB
  const size_t szW  = (size_t)D_ * D_ * sizeof(bf16); // 2 MB
  bf16* qb  = alloc(szXD);
  bf16* kb  = alloc(szXD);
  bf16* vb  = alloc(szXD);
  bf16* wqb = alloc(szW);
  bf16* wkb = alloc(szW);
  bf16* wvb = alloc(szW);
  bf16* wob = alloc(szW);
  bf16* Qh  = alloc(szXD);
  bf16* Kh  = alloc(szXD);
  bf16* Vh  = alloc(szXD);
  bf16* xb  = alloc(szXD);

  const int nXD = M_ * D_, nW = D_ * D_;
  cvt_f32_to_bf16<<<1024, 256, 0, stream>>>(q, qb, nXD);
  cvt_f32_to_bf16<<<1024, 256, 0, stream>>>(k, kb, nXD);
  cvt_f32_to_bf16<<<1024, 256, 0, stream>>>(v, vb, nXD);
  cvt_f32_to_bf16<<<512, 256, 0, stream>>>(Wq, wqb, nW);
  cvt_f32_to_bf16<<<512, 256, 0, stream>>>(Wk, wkb, nW);
  cvt_f32_to_bf16<<<512, 256, 0, stream>>>(Wv, wvb, nW);
  cvt_f32_to_bf16<<<512, 256, 0, stream>>>(Wo, wob, nW);

  const dim3 gg(D_ / 128, M_ / 128); // (8, 32), 256 threads
  gemm_bf16<<<gg, 256, 0, stream>>>(qb, wqb, bq, nullptr, Qh, M_, D_, D_, 1);
  gemm_bf16<<<gg, 256, 0, stream>>>(kb, wkb, bk, nullptr, Kh, M_, D_, D_, 1);
  gemm_bf16<<<gg, 256, 0, stream>>>(vb, wvb, bv, nullptr, Vh, M_, D_, D_, 1);

  flash_attn<<<dim3(S_ / 128, B_ * H_), 256, 0, stream>>>(Qh, Kh, Vh, mask, xb);

  gemm_bf16<<<gg, 256, 0, stream>>>(xb, wob, bo, out, nullptr, M_, D_, D_, 0);
}
